// DecoderLayer_71141838291441
// MI455X (gfx1250) — compile-verified, matrix operand order checked by reading, not run on hardware
//
#include <hip/hip_runtime.h>
#include <math.h>
#include <stdint.h>

// ---------------- problem constants ----------------
#define B_   2
#define S_   2048
#define D_   1024
#define T_   (B_ * S_)          // 4096 tokens
#define NQ_  16
#define NKV_ 4
#define HD_  64
#define NREP_ (NQ_ / NKV_)
#define E_   8
#define DFF_ 4096
#define KVD_ (NKV_ * HD_)       // 256
#define EPS_ 1e-8f

// ---------------- vector types ----------------
typedef __attribute__((ext_vector_type(16))) __bf16 v16bf;
typedef __attribute__((ext_vector_type(8)))  __bf16 v8bf;
typedef __attribute__((ext_vector_type(8)))  float  v8f;
typedef __attribute__((ext_vector_type(4)))  unsigned int u32x4;
typedef __attribute__((ext_vector_type(4)))  int    i32x4;
typedef __attribute__((ext_vector_type(8)))  int    i32x8;

union FragU { v16bf v; v8bf h[2]; };

__device__ inline v8f vzero8() {
  v8f z = {0.f,0.f,0.f,0.f,0.f,0.f,0.f,0.f};
  return z;
}

// Load a 16x32 bf16 operand fragment (A layout; B uses the same math with
// "row" = output column, reading an N-major [n][k] weight).
__device__ inline v16bf load_frag(const __bf16* base, int stride) {
  const int lane = threadIdx.x & 31;
  const int row  = lane & 15;
  const int kh   = lane >> 4;
  const __bf16* p = base + (long long)row * stride + kh * 8;
  FragU u;
  u.h[0] = *(const v8bf*)(p);
  u.h[1] = *(const v8bf*)(p + 16);
  return u.v;
}

__device__ inline v8f wmma_bf16(v16bf a, v16bf b, v8f c) {
  return __builtin_amdgcn_wmma_f32_16x16x32_bf16(false, a, false, b,
                                                 (short)0, c, false, false);
}

// ---------------- TDM: 2D tile load (bf16 elements) to LDS ----------------
// D# per CDNA5 ISA ch.8: group0 = {count/type/addr}, group1 = {dims/strides}.
// Units: tensor/tile dims and stride0 in elements (data_size = 2 bytes).
__device__ inline void tdm_load_2d(unsigned lds_off, unsigned long long gaddr,
                                   unsigned tensor_d0, unsigned tensor_d1,
                                   unsigned tile_d0, unsigned tile_d1,
                                   unsigned long long stride0) {
  u32x4 g0;
  g0[0] = 1u;                                            // count=1, user mode
  g0[1] = lds_off;                                       // lds_addr (bytes)
  g0[2] = (unsigned)gaddr;                               // global_addr[31:0]
  g0[3] = (unsigned)((gaddr >> 32) & 0x1FFFFFFull) | 0x80000000u; // [56:32]|type=2
  i32x8 g1;
  g1[0] = 0x00010000;                                    // data_size = 2 bytes
  g1[1] = (int)((tensor_d0 & 0xFFFFu) << 16);            // tensor_dim0 lo16
  g1[2] = (int)((tensor_d0 >> 16) | ((tensor_d1 & 0xFFFFu) << 16));
  g1[3] = (int)((tensor_d1 >> 16) | (tile_d0 << 16));    // tile_dim0
  g1[4] = (int)(tile_d1 & 0xFFFFu);                      // tile_dim1 (tile_dim2=0)
  g1[5] = (int)(unsigned)(stride0 & 0xFFFFFFFFull);      // dim0_stride lo32
  g1[6] = (int)(unsigned)((stride0 >> 32) & 0xFFFFull);  // dim0_stride hi16
  g1[7] = 0;
  i32x4 z4 = {0,0,0,0};
#if defined(__clang_major__) && (__clang_major__ >= 23)
  i32x8 z8 = {0,0,0,0,0,0,0,0};
  __builtin_amdgcn_tensor_load_to_lds(g0, g1, z4, z4, z8, 0);
#else
  __builtin_amdgcn_tensor_load_to_lds(g0, g1, z4, z4, 0);
#endif
}

// ---------------- weight pack: fp32 [b][K][N] -> bf16 [b][N][K] ----------------
__global__ void pack_w_kernel(const float* __restrict__ src, __bf16* __restrict__ dst,
                              int K, int N) {
  const long long b = blockIdx.z;
  const float* s = src + b * (long long)K * N;
  __bf16*      d = dst + b * (long long)N * K;
  long long idx = (long long)blockIdx.x * blockDim.x + threadIdx.x;
  if (idx >= (long long)N * K) return;
  int n = (int)(idx / K);
  int k = (int)(idx % K);
  d[idx] = (__bf16)s[(long long)k * N + n];
}

// ---------------- RMSNorm: f32 [T][D] -> bf16 [T][D] ----------------
__global__ __launch_bounds__(256) void rmsnorm_kernel(const float* __restrict__ x,
                                                      const float* __restrict__ w,
                                                      __bf16* __restrict__ out) {
  const int t = blockIdx.x;
  __shared__ float red[256];
  float s = 0.f;
  for (int i = threadIdx.x; i < D_; i += 256) {
    float v = x[(long long)t * D_ + i];
    s += v * v;
  }
  red[threadIdx.x] = s;
  __syncthreads();
  for (int st = 128; st > 0; st >>= 1) {
    if (threadIdx.x < st) red[threadIdx.x] += red[threadIdx.x + st];
    __syncthreads();
  }
  const float inv = rsqrtf(red[0] / (float)D_ + EPS_);
  for (int i = threadIdx.x; i < D_; i += 256)
    out[(long long)t * D_ + i] = (__bf16)(x[(long long)t * D_ + i] * inv * w[i]);
}

// ---------------- generic bf16 GEMM: C[M][N] = A[M][K] @ Wp[N][K] ----------------
// mode 0: store bf16 to outp.  mode 1: v = acc + resid; store f32 to outp and out2.
__global__ __launch_bounds__(128) void gemm_bf16_kernel(const __bf16* __restrict__ A,
                                                        const __bf16* __restrict__ Wp,
                                                        void* __restrict__ outp,
                                                        const float* __restrict__ resid,
                                                        float* __restrict__ out2,
                                                        int M, int N, int K, int mode) {
  const int w = threadIdx.x >> 5, lane = threadIdx.x & 31;
  const int m0 = blockIdx.y * 64 + w * 16;
  const int n0 = blockIdx.x * 64;
  (void)M;
  v8f c[4];
  for (int j = 0; j < 4; ++j) c[j] = vzero8();
  const __bf16* Ap = A + (long long)m0 * K;
  for (int kk = 0; kk < K; kk += 32) {
    __builtin_prefetch(Ap + kk + 128, 0, 3);          // global_prefetch_b8
    v16bf a = load_frag(Ap + kk, K);
#pragma unroll
    for (int j = 0; j < 4; ++j) {
      const __bf16* Bp = Wp + (long long)(n0 + 16 * j) * K + kk;
      __builtin_prefetch(Bp + 128, 0, 3);
      v16bf b = load_frag(Bp, K);
      c[j] = wmma_bf16(a, b, c[j]);
    }
  }
  const int col = lane & 15, kh = lane >> 4;
  for (int j = 0; j < 4; ++j)
    for (int r = 0; r < 8; ++r) {
      long long idx = (long long)(m0 + r + 8 * kh) * N + n0 + 16 * j + col;
      if (mode == 0) {
        ((__bf16*)outp)[idx] = (__bf16)c[j][r];
      } else {
        float v = c[j][r] + resid[idx];
        ((float*)outp)[idx] = v;
        out2[idx] = v;
      }
    }
}

// ---------------- RoPE (in-place, interleaved pairs) ----------------
__global__ void rope_kernel(__bf16* __restrict__ X, int nheads) {
  long long tid = (long long)blockIdx.x * blockDim.x + threadIdx.x;
  long long total = (long long)T_ * nheads * (HD_ / 2);
  if (tid >= total) return;
  const int  i   = (int)(tid & 31);          // pair index (HD/2 == 32)
  long long  th  = tid >> 5;                 // token*nheads + head
  long long  tok = th / nheads;
  const int  s   = (int)(tok & (S_ - 1));    // position in sequence
  const float inv = __expf(-((float)(2 * i) / (float)HD_) * 9.210340371976184f); // ln(1e4)
  const float ang = (float)s * inv;
  const float sn = __sinf(ang), cs = __cosf(ang);
  long long base = th * HD_ + 2 * i;
  float e = (float)X[base], o = (float)X[base + 1];
  X[base]     = (__bf16)(e * cs - o * sn);
  X[base + 1] = (__bf16)(o * cs + e * sn);
}

// ---------------- flash attention (causal, GQA), WMMA bf16 + TDM K staging ---
// grid: (S/64, NQ, B); block 128 (4 waves, each wave owns a 16-query tile).
// K tiles are DMA'd into LDS by the Tensor Data Mover, double-buffered so the
// DMA of tile kt+1 overlaps WMMA compute on tile kt.
__global__ __launch_bounds__(128) void flash_attn_kernel(const __bf16* __restrict__ Q,
                                                         const __bf16* __restrict__ Kb,
                                                         const __bf16* __restrict__ Vb,
                                                         __bf16* __restrict__ O) {
  const int b = blockIdx.z, h = blockIdx.y;
  const int w = threadIdx.x >> 5, lane = threadIdx.x & 31;
  const int q0 = (blockIdx.x * 4 + w) * 16;
  const int kvh = h >> 2;                                   // NREP = 4
  __shared__ __bf16 Kl[2][32 * 64];                         // double-buffered [key][d]
  __shared__ __bf16 Vt[64 * 32];                            // [d][key]
  __shared__ __bf16 Pl[4][16 * 32];                         // per-wave P

  const __bf16* Qp = Q + ((long long)(b * S_ + q0)) * D_ + h * HD_;
  const v16bf qa0 = load_frag(Qp, D_);
  const v16bf qa1 = load_frag(Qp + 32, D_);

  float m[8], l[8];
  v8f o[4];
  for (int r = 0; r < 8; ++r) { m[r] = -3.0e38f; l[r] = 0.f; }
  for (int j = 0; j < 4; ++j) o[j] = vzero8();

  const int ktmax_blk = blockIdx.x * 2 + 1;                 // block causal bound
  const int ktmax_w   = (q0 + 15) >> 5;                     // wave causal bound
  const long long kvbase = (long long)(b * S_) * KVD_ + kvh * HD_;
  const unsigned long long kbyte = (unsigned long long)(size_t)Kb + 2ull * (unsigned long long)kvbase;

  const int col = lane & 15, kh = lane >> 4;

  // preload K tile 0 via TDM (wave 0 issues; TENSORcnt tracks completion)
  if (w == 0) {
    tdm_load_2d((unsigned)(size_t)&Kl[0][0], kbyte,
                /*tensor_d0*/HD_, /*tensor_d1*/32, /*tile_d0*/HD_, /*tile_d1*/32,
                /*stride0*/KVD_);
  }

  for (int kt = 0; kt <= ktmax_blk; ++kt) {
    { // cooperative staging of V (transposed) for this tile
      const int tid = threadIdx.x;
      const int key = tid >> 2;
      const int dc  = (tid & 3) * 16;
      const __bf16* vs = Vb + kvbase + (long long)(kt * 32 + key) * KVD_ + dc;
      for (int d = 0; d < 16; ++d) Vt[(dc + d) * 32 + key] = vs[d];
    }
    // wave 0: kick off DMA of the next K tile, then wait for the current one
    if (w == 0) {
      if (kt + 1 <= ktmax_blk) {
        tdm_load_2d((unsigned)(size_t)&Kl[(kt + 1) & 1][0],
                    kbyte + 2ull * (unsigned long long)(kt + 1) * 32ull * KVD_,
                    HD_, 32, HD_, 32, KVD_);
        __builtin_amdgcn_s_wait_tensorcnt(1);
      } else {
        __builtin_amdgcn_s_wait_tensorcnt(0);
      }
    }
    __syncthreads();

    if (kt <= ktmax_w) {
      const __bf16* Kc = &Kl[kt & 1][0];
      v8f c0 = vzero8(), c1 = vzero8();
      v16bf b00 = load_frag(Kc, 64);
      v16bf b01 = load_frag(Kc + 32, 64);
      v16bf b10 = load_frag(Kc + 16 * 64, 64);
      v16bf b11 = load_frag(Kc + 16 * 64 + 32, 64);
      c0 = wmma_bf16(qa0, b00, c0);
      c0 = wmma_bf16(qa1, b01, c0);
      c1 = wmma_bf16(qa0, b10, c1);
      c1 = wmma_bf16(qa1, b11, c1);

      const float scale = 0.125f;                           // 1/sqrt(64)
      const int key0 = kt * 32 + col, key1 = kt * 32 + 16 + col;
#pragma unroll
      for (int r = 0; r < 8; ++r) {
        const int q = q0 + r + 8 * kh;
        float a0 = (key0 <= q) ? c0[r] * scale : -3.0e38f;
        float a1 = (key1 <= q) ? c1[r] * scale : -3.0e38f;
        float rm = fmaxf(a0, a1);
        for (int msk = 8; msk > 0; msk >>= 1)
          rm = fmaxf(rm, __shfl_xor(rm, msk, 32));
        const float nm = fmaxf(m[r], rm);
        float p0 = (a0 > -1.0e37f) ? __expf(a0 - nm) : 0.f;
        float p1 = (a1 > -1.0e37f) ? __expf(a1 - nm) : 0.f;
        float rs = p0 + p1;
        for (int msk = 8; msk > 0; msk >>= 1)
          rs += __shfl_xor(rs, msk, 32);
        const float f = __expf(m[r] - nm);
        l[r] = l[r] * f + rs;
        m[r] = nm;
        o[0][r] *= f; o[1][r] *= f; o[2][r] *= f; o[3][r] *= f;
        const int row = r + 8 * kh;
        Pl[w][row * 32 + col]      = (__bf16)p0;
        Pl[w][row * 32 + 16 + col] = (__bf16)p1;
      }
      v16bf pa = load_frag(&Pl[w][0], 32);
#pragma unroll
      for (int j = 0; j < 4; ++j) {
        v16bf vb = load_frag(&Vt[(16 * j) * 32], 32);
        o[j] = wmma_bf16(pa, vb, o[j]);
      }
    }
    __syncthreads();
  }

  for (int j = 0; j < 4; ++j)
    for (int r = 0; r < 8; ++r) {
      const int q = q0 + r + 8 * kh;
      O[((long long)(b * S_ + q)) * D_ + h * HD_ + 16 * j + col] =
          (__bf16)(o[j][r] / l[r]);
    }
}

// ---------------- router: noisy top-2 softmax + importance ----------------
__global__ __launch_bounds__(256) void router_kernel(const __bf16* __restrict__ h2,
                                                     const float* __restrict__ gate_w,
                                                     const float* __restrict__ gate_b,
                                                     const float* __restrict__ noise_w,
                                                     const float* __restrict__ noise_b,
                                                     const float* __restrict__ rnoise,
                                                     float* __restrict__ scores,
                                                     float* __restrict__ importance) {
  const int t = blockIdx.x * 8 + (threadIdx.x >> 5);
  const int lane = threadIdx.x & 31;
  float acc[E_];
  for (int e = 0; e < E_; ++e) acc[e] = 0.f;
  for (int k = lane; k < D_; k += 32) {
    const float hv = (float)h2[(long long)t * D_ + k];
#pragma unroll
    for (int e = 0; e < E_; ++e)
      acc[e] += hv * (gate_w[k * E_ + e] + noise_w[k * E_ + e]);
  }
#pragma unroll
  for (int e = 0; e < E_; ++e)
    for (int msk = 16; msk > 0; msk >>= 1)
      acc[e] += __shfl_xor(acc[e], msk, 32);
  if (lane == 0) {
    float lg[E_];
    for (int e = 0; e < E_; ++e)
      lg[e] = acc[e] + gate_b[e] + noise_b[e] + rnoise[(long long)t * E_ + e];
    int i1 = 0;
    for (int e = 1; e < E_; ++e) if (lg[e] > lg[i1]) i1 = e;
    int i2 = -1;
    for (int e = 0; e < E_; ++e)
      if (e != i1 && (i2 < 0 || lg[e] > lg[i2])) i2 = e;
    const float mx = fmaxf(lg[i1], lg[i2]);
    const float e1 = __expf(lg[i1] - mx), e2 = __expf(lg[i2] - mx);
    const float s1 = e1 / (e1 + e2), s2 = e2 / (e1 + e2);
    for (int e = 0; e < E_; ++e) scores[(long long)t * E_ + e] = 0.f;
    scores[(long long)t * E_ + i1] = s1;
    scores[(long long)t * E_ + i2] = s2;
    atomicAdd(&importance[i1], s1 / (float)T_);
    atomicAdd(&importance[i2], s2 / (float)T_);
  }
}

// ---------------- fused MoE: per (expert, 16-token tile) ----------------
// 8 waves; TDM stages the 16x1024 activation tile to LDS once, then
// GEMM1 -> SiLU -> LDS(hh) -> GEMM2 (regs) -> score-weighted atomicAdd.
#define FFC 512
__global__ __launch_bounds__(256) void moe_kernel(const __bf16* __restrict__ h2,
                                                  const __bf16* __restrict__ ew1p,
                                                  const __bf16* __restrict__ ew2p,
                                                  const float* __restrict__ eb1,
                                                  const float* __restrict__ eb2,
                                                  const float* __restrict__ scores,
                                                  float* __restrict__ out) {
  const int e  = blockIdx.y;
  const int t0 = blockIdx.x * 16;
  __shared__ float  s_sc[16];
  __shared__ __bf16 Al[16 * D_];                             // 32 KB activation tile
  __shared__ __bf16 hh[16 * FFC];                            // 16 KB
  const int tid = threadIdx.x;
  if (tid < 16) s_sc[tid] = scores[(long long)(t0 + tid) * E_ + e];
  __syncthreads();
  bool any = false;
  for (int i = 0; i < 16; ++i) any |= (s_sc[i] > 0.f);
  if (!any) return;                                          // uniform exit

  const int w = tid >> 5, lane = tid & 31;
  // TDM: DMA the 16xD activation tile into LDS (wave 0 issues)
  if (w == 0) {
    tdm_load_2d((unsigned)(size_t)&Al[0],
                (unsigned long long)(size_t)(h2 + (long long)t0 * D_),
                /*tensor_d0*/D_, /*tensor_d1*/16, /*tile_d0*/D_, /*tile_d1*/16,
                /*stride0*/D_);
    __builtin_amdgcn_s_wait_tensorcnt(0);
  }
  __syncthreads();

  const __bf16* W1 = ew1p + (long long)e * DFF_ * D_;        // [ff][d]
  const __bf16* W2 = ew2p + (long long)e * D_ * DFF_;        // [d][ff]

  v8f acc2[8];
  for (int j = 0; j < 8; ++j) acc2[j] = vzero8();

  const int colL = lane & 15, khL = lane >> 4;
  for (int ff0 = 0; ff0 < DFF_; ff0 += FFC) {
    // GEMM1: hh[16][FFC] chunk; wave covers 64 ff-columns (4 tiles)
    v8f acc1[4];
    for (int j = 0; j < 4; ++j) acc1[j] = vzero8();
    for (int kk = 0; kk < D_; kk += 32) {
      v16bf a = load_frag(&Al[kk], D_);
#pragma unroll
      for (int j = 0; j < 4; ++j) {
        const __bf16* Bp = W1 + (long long)(ff0 + w * 64 + j * 16) * D_ + kk;
        __builtin_prefetch(Bp + 128, 0, 3);
        v16bf b = load_frag(Bp, D_);
        acc1[j] = wmma_bf16(a, b, acc1[j]);
      }
    }
#pragma unroll
    for (int j = 0; j < 4; ++j) {
      const int n = w * 64 + j * 16 + colL;
      const float bias = eb1[(long long)e * DFF_ + ff0 + n];
      for (int r = 0; r < 8; ++r) {
        float v = acc1[j][r] + bias;
        v = v / (1.f + __expf(-v));                          // SiLU
        hh[(r + 8 * khL) * FFC + n] = (__bf16)v;
      }
    }
    __syncthreads();
    // GEMM2: acc2 over D-columns n = w*128 + 16j
    for (int kk = 0; kk < FFC; kk += 32) {
      v16bf a = load_frag(&hh[kk], FFC);
#pragma unroll
      for (int j = 0; j < 8; ++j) {
        const __bf16* Bp = W2 + (long long)(w * 128 + j * 16) * DFF_ + ff0 + kk;
        __builtin_prefetch(Bp + 128, 0, 3);
        v16bf b = load_frag(Bp, DFF_);
        acc2[j] = wmma_bf16(a, b, acc2[j]);
      }
    }
    __syncthreads();
  }
#pragma unroll
  for (int j = 0; j < 8; ++j) {
    const int n = w * 128 + j * 16 + colL;
    const float bias = eb2[(long long)e * D_ + n];
    for (int r = 0; r < 8; ++r) {
      const int row = r + 8 * khL;
      const float sc = s_sc[row];
      if (sc > 0.f)
        atomicAdd(&out[(long long)(t0 + row) * D_ + n], sc * (acc2[j][r] + bias));
    }
  }
}

// ---------------- small utility kernels ----------------
__global__ void zero_kernel(float* p, int n) {
  int i = blockIdx.x * blockDim.x + threadIdx.x;
  if (i < n) p[i] = 0.f;
}

__global__ void aux_kernel(const float* __restrict__ imp, float* __restrict__ out_aux) {
  if (threadIdx.x == 0 && blockIdx.x == 0) {
    const float u = 1.f / (float)E_;
    float a = 0.f;
    for (int e = 0; e < E_; ++e) a += u * (logf(u) - logf(imp[e] + EPS_));
    *out_aux = a;
  }
}

// ---------------- host orchestration ----------------
extern "C" void kernel_launch(void* const* d_in, const int* in_sizes, int n_in,
                              void* d_out, int out_size, void* d_ws, size_t ws_size,
                              hipStream_t stream) {
  (void)in_sizes; (void)n_in; (void)out_size; (void)ws_size;

  const float* x       = (const float*)d_in[0];
  const float* rnoise  = (const float*)d_in[1];
  const float* n1w     = (const float*)d_in[2];
  const float* n2w     = (const float*)d_in[3];
  const float* wq      = (const float*)d_in[4];
  const float* wk      = (const float*)d_in[5];
  const float* wv      = (const float*)d_in[6];
  const float* wo      = (const float*)d_in[7];
  const float* gate_w  = (const float*)d_in[8];
  const float* gate_b  = (const float*)d_in[9];
  const float* noise_w = (const float*)d_in[10];
  const float* noise_b = (const float*)d_in[11];
  const float* ew1     = (const float*)d_in[12];
  const float* eb1     = (const float*)d_in[13];
  const float* ew2     = (const float*)d_in[14];
  const float* eb2     = (const float*)d_in[15];
  float* out = (float*)d_out;

  // workspace carve-out
  char* p = (char*)d_ws;
  auto carve = [&](size_t bytes) {
    char* r = p;
    p += (bytes + 255) & ~(size_t)255;
    return (void*)r;
  };
  __bf16* h1     = (__bf16*)carve((size_t)T_ * D_ * 2);
  __bf16* qb     = (__bf16*)carve((size_t)T_ * D_ * 2);
  __bf16* kb     = (__bf16*)carve((size_t)T_ * KVD_ * 2);
  __bf16* vb     = (__bf16*)carve((size_t)T_ * KVD_ * 2);
  __bf16* attn   = (__bf16*)carve((size_t)T_ * D_ * 2);
  float*  x_attn = (float*) carve((size_t)T_ * D_ * 4);
  __bf16* h2     = (__bf16*)carve((size_t)T_ * D_ * 2);
  float*  scores = (float*) carve((size_t)T_ * E_ * 4);
  float*  imp    = (float*) carve((size_t)E_ * 4);
  __bf16* wq_p   = (__bf16*)carve((size_t)D_ * D_ * 2);
  __bf16* wk_p   = (__bf16*)carve((size_t)KVD_ * D_ * 2);
  __bf16* wv_p   = (__bf16*)carve((size_t)KVD_ * D_ * 2);
  __bf16* wo_p   = (__bf16*)carve((size_t)D_ * D_ * 2);
  __bf16* ew1_p  = (__bf16*)carve((size_t)E_ * DFF_ * D_ * 2);
  __bf16* ew2_p  = (__bf16*)carve((size_t)E_ * D_ * DFF_ * 2);

  auto cdiv = [](long long a, long long b) { return (unsigned)((a + b - 1) / b); };

  // 1) pack weights to bf16, N-major [n][k]
  pack_w_kernel<<<dim3(cdiv((long long)D_ * D_, 256), 1, 1), 256, 0, stream>>>(wq, wq_p, D_, D_);
  pack_w_kernel<<<dim3(cdiv((long long)KVD_ * D_, 256), 1, 1), 256, 0, stream>>>(wk, wk_p, D_, KVD_);
  pack_w_kernel<<<dim3(cdiv((long long)KVD_ * D_, 256), 1, 1), 256, 0, stream>>>(wv, wv_p, D_, KVD_);
  pack_w_kernel<<<dim3(cdiv((long long)D_ * D_, 256), 1, 1), 256, 0, stream>>>(wo, wo_p, D_, D_);
  pack_w_kernel<<<dim3(cdiv((long long)DFF_ * D_, 256), 1, E_), 256, 0, stream>>>(ew1, ew1_p, D_, DFF_);
  pack_w_kernel<<<dim3(cdiv((long long)D_ * DFF_, 256), 1, E_), 256, 0, stream>>>(ew2, ew2_p, DFF_, D_);

  // 2) rmsnorm1
  rmsnorm_kernel<<<T_, 256, 0, stream>>>(x, n1w, h1);

  // 3) QKV projections (WMMA)
  gemm_bf16_kernel<<<dim3(D_ / 64, T_ / 64), 128, 0, stream>>>(h1, wq_p, qb, nullptr, nullptr, T_, D_, D_, 0);
  gemm_bf16_kernel<<<dim3(KVD_ / 64, T_ / 64), 128, 0, stream>>>(h1, wk_p, kb, nullptr, nullptr, T_, KVD_, D_, 0);
  gemm_bf16_kernel<<<dim3(KVD_ / 64, T_ / 64), 128, 0, stream>>>(h1, wv_p, vb, nullptr, nullptr, T_, KVD_, D_, 0);

  // 4) RoPE on Q and K
  {
    long long tq = (long long)T_ * NQ_ * (HD_ / 2);
    rope_kernel<<<cdiv(tq, 256), 256, 0, stream>>>(qb, NQ_);
    long long tk = (long long)T_ * NKV_ * (HD_ / 2);
    rope_kernel<<<cdiv(tk, 256), 256, 0, stream>>>(kb, NKV_);
  }

  // 5) flash attention (TDM-staged K tiles)
  flash_attn_kernel<<<dim3(S_ / 64, NQ_, B_), 128, 0, stream>>>(qb, kb, vb, attn);

  // 6) output projection + residual -> x_attn (ws) and d_out
  gemm_bf16_kernel<<<dim3(D_ / 64, T_ / 64), 128, 0, stream>>>(attn, wo_p, x_attn, x, out, T_, D_, D_, 1);

  // 7) rmsnorm2
  rmsnorm_kernel<<<T_, 256, 0, stream>>>(x_attn, n2w, h2);

  // 8) router
  zero_kernel<<<1, 32, 0, stream>>>(imp, E_);
  router_kernel<<<T_ / 8, 256, 0, stream>>>(h2, gate_w, gate_b, noise_w, noise_b,
                                            rnoise, scores, imp);

  // 9) fused MoE (TDM-staged activations, atomicAdd onto d_out)
  moe_kernel<<<dim3(T_ / 16, E_), 256, 0, stream>>>(h2, ew1_p, ew2_p, eb1, eb2, scores, out);

  // 10) aux loss scalar
  aux_kernel<<<1, 32, 0, stream>>>(imp, out + (size_t)T_ * D_);
}
